// VectorQuantizer_72679436583362
// MI455X (gfx1250) — matrix operand order repaired and checked by reading, not compiled, not run on hardware
//
#include <hip/hip_runtime.h>
#include <hip/hip_bf16.h>
#include <math.h>

typedef _Float16 v16h __attribute__((ext_vector_type(16)));
typedef _Float16 v8h  __attribute__((ext_vector_type(8)));
typedef float    v8f  __attribute__((ext_vector_type(8)));
typedef int      ivec4 __attribute__((vector_size(16)));   // matches builtin's V4i

#define BB 16
#define DD 256
#define TT 4096
#define KK 1024
#define NN (BB * TT)               // 65536 rows
#define QELEMS ((size_t)NN * DD)   // 16777216

// CDNA5 async global->LDS path (guarded: falls back to sync staging if the
// toolchain doesn't declare the builtins).
#if defined(__AMDGCN__) && __has_builtin(__builtin_amdgcn_global_load_async_to_lds_b128)
#define HAVE_ASYNC_LDS 1
#else
#define HAVE_ASYNC_LDS 0
#endif

#if defined(__AMDGCN__) && __has_builtin(__builtin_amdgcn_s_wait_asynccnt)
#define WAIT_ASYNC(n) __builtin_amdgcn_s_wait_asynccnt(n)
#elif defined(__AMDGCN__)
#define WAIT_ASYNC(n) asm volatile("s_wait_asynccnt %0" ::"i"(n) : "memory")
#else
#define WAIT_ASYNC(n)
#endif

// ---------------------------------------------------------------------------
// Kernel 1: L2-normalize embeddings [1024 x 256] f32 -> f16 table in ws.
// One wave per embedding row (8 f32 per lane). Also zeroes loss accumulator.
// ---------------------------------------------------------------------------
__global__ __launch_bounds__(256) void vq_prep(const float* __restrict__ emb,
                                               _Float16* __restrict__ enorm,
                                               float* __restrict__ acc) {
    int tid  = threadIdx.x;
    int wave = tid >> 5;
    int lane = tid & 31;
    int row  = blockIdx.x * 8 + wave;

    const float* src = emb + (size_t)row * DD + lane * 8;
    float4 p0 = *(const float4*)(src);
    float4 p1 = *(const float4*)(src + 4);
    float s = p0.x * p0.x + p0.y * p0.y + p0.z * p0.z + p0.w * p0.w +
              p1.x * p1.x + p1.y * p1.y + p1.z * p1.z + p1.w * p1.w;
#pragma unroll
    for (int msk = 1; msk < 32; msk <<= 1) s += __shfl_xor(s, msk, 32);
    float rn = 1.0f / fmaxf(sqrtf(s), 1e-12f);

    v8h o;
    o[0] = (_Float16)(p0.x * rn); o[1] = (_Float16)(p0.y * rn);
    o[2] = (_Float16)(p0.z * rn); o[3] = (_Float16)(p0.w * rn);
    o[4] = (_Float16)(p1.x * rn); o[5] = (_Float16)(p1.y * rn);
    o[6] = (_Float16)(p1.z * rn); o[7] = (_Float16)(p1.w * rn);
    *(v8h*)(enorm + (size_t)row * DD + lane * 8) = o;

    if (blockIdx.x == 0 && tid == 0) *acc = 0.0f;
}

// ---------------------------------------------------------------------------
// Kernel 2: main VQ kernel. Block = 64 threads (2 waves). Each wave handles a
// 16-row tile of N. Similarity GEMM [16x256]x[256x1024] via WMMA f16 with a
// running argmax in registers; then gather + transpose-store + loss.
// B tiles are double-buffered in LDS via async global->LDS DMA when available.
// ---------------------------------------------------------------------------
#define XPITCH 260   // 16B-aligned, bank-conflict-free pitch for f32 rows
#define BPITCH 264   // same for f16 rows

__global__ __launch_bounds__(64) void vq_main(const float* __restrict__ inp,
                                              const float* __restrict__ emb,
                                              const _Float16* __restrict__ enorm,
                                              float* __restrict__ outq,
                                              float* __restrict__ outidx,
                                              float* __restrict__ acc) {
    __shared__ float    xt[2][16][XPITCH];   // raw x tiles, one per wave
    __shared__ _Float16 bt[2][16][BPITCH];   // double-buffered 16-embedding tile

    int tid  = threadIdx.x;
    int w    = tid >> 5;        // wave in block
    int lane = tid & 31;
    int m    = lane & 15;       // row within 16-row tile / column within C tile
    int g    = lane >> 4;       // lane half

    int n0 = blockIdx.x * 32 + w * 16;   // first flat row of this wave's tile
    int b  = n0 >> 12;                   // T = 4096
    int t0 = n0 & (TT - 1);

#if HAVE_ASYNC_LDS
    // Issue one 8 KB B tile: 16 rows x 512 B; each wave DMAs 8 rows, one
    // async b128 per row (32 lanes x 16 B). Tracked by ASYNCcnt.
    auto issue_tile = [&](int buf, int nt2) {
#pragma unroll
        for (int r = 0; r < 8; ++r) {
            int row = r * 2 + w;
            const char* gp = (const char*)enorm +
                             ((size_t)nt2 * 16 * DD + row * DD) * sizeof(_Float16) +
                             lane * 16;
            __builtin_amdgcn_global_load_async_to_lds_b128(
                (__attribute__((address_space(1))) ivec4*)gp,
                (__attribute__((address_space(3))) ivec4*)&bt[buf][row][lane * 8],
                0, 0);
        }
    };
#endif

    // ---- Phase A: stage x tile (coalesced along t), row sumsq ----
#if HAVE_ASYNC_LDS
    issue_tile(0, 0);            // overlap first B tile DMA with Phase A
#endif
    const float* xin = inp + (size_t)b * DD * TT + t0 + m;
    float ss = 0.0f;
#pragma unroll 4
    for (int i = 0; i < 128; ++i) {
        int d = 2 * i + g;                       // lane pair (L, L+16) covers all d
        float v = xin[(size_t)d * TT];
        xt[w][m][d] = v;
        ss += v * v;
    }
    ss += __shfl_xor(ss, 16, 32);                // combine parity halves of row m
    float rn = 1.0f / fmaxf(sqrtf(ss), 1e-12f);

    // ---- Pack A operand: 8 chunks of 16x32 f16 (ISA §7.12.2 layout) ----
    // lane holds row m; element i<8 -> krel = g*8+i ; i>=8 -> krel = 16+g*8+(i-8)
    v16h a[8];
#pragma unroll
    for (int kc = 0; kc < 8; ++kc) {
        const float* xr = &xt[w][m][0];
        int base = kc * 32 + g * 8;
        float4 r0 = *(const float4*)(xr + base);
        float4 r1 = *(const float4*)(xr + base + 4);
        float4 r2 = *(const float4*)(xr + base + 16);
        float4 r3 = *(const float4*)(xr + base + 20);
        a[kc][0]  = (_Float16)(r0.x * rn); a[kc][1]  = (_Float16)(r0.y * rn);
        a[kc][2]  = (_Float16)(r0.z * rn); a[kc][3]  = (_Float16)(r0.w * rn);
        a[kc][4]  = (_Float16)(r1.x * rn); a[kc][5]  = (_Float16)(r1.y * rn);
        a[kc][6]  = (_Float16)(r1.z * rn); a[kc][7]  = (_Float16)(r1.w * rn);
        a[kc][8]  = (_Float16)(r2.x * rn); a[kc][9]  = (_Float16)(r2.y * rn);
        a[kc][10] = (_Float16)(r2.z * rn); a[kc][11] = (_Float16)(r2.w * rn);
        a[kc][12] = (_Float16)(r3.x * rn); a[kc][13] = (_Float16)(r3.y * rn);
        a[kc][14] = (_Float16)(r3.z * rn); a[kc][15] = (_Float16)(r3.w * rn);
    }

    // ---- Phase B: sweep 64 embedding tiles, WMMA + running argmax ----
    float maxv[8]; int maxi[8];
#pragma unroll
    for (int r = 0; r < 8; ++r) { maxv[r] = -3.4e38f; maxi[r] = 0; }

    for (int nt = 0; nt < 64; ++nt) {
        int cur = nt & 1;
#if HAVE_ASYNC_LDS
        // Prefetch next tile into the other buffer, then wait for current:
        // this wave has issued 8 newer (next-tile) ops; ASYNCcnt completes
        // in order, so cnt<=8 means the current tile's DMA is done.
        if (nt < 63) {
            issue_tile(cur ^ 1, nt + 1);
            WAIT_ASYNC(8);
        } else {
            WAIT_ASYNC(0);
        }
#else
        // Fallback: cooperative sync stage of the 8 KB tile through VGPRs.
        const v8h* gsrc = (const v8h*)(enorm + (size_t)nt * 16 * DD);
#pragma unroll
        for (int r = 0; r < 8; ++r) {
            int row = r * 2 + w;
            v8h v = gsrc[row * 32 + lane];
            *(v8h*)&bt[cur][row][lane * 8] = v;
        }
        if (nt < 63) {   // prefetch next tile -> global_prefetch_b8
            const char* nxt = (const char*)(enorm + (size_t)(nt + 1) * 16 * DD);
            __builtin_prefetch(nxt + tid * 128, 0, 1);
        }
#endif
        __syncthreads();   // all waves: current tile resident in bt[cur]

        v8f c = {0.f, 0.f, 0.f, 0.f, 0.f, 0.f, 0.f, 0.f};
#pragma unroll
        for (int kc = 0; kc < 8; ++kc) {
            // B operand: lane = column m, element e<8 -> k = kc*32 + g*8 + e ; e>=8 -> k = kc*32 + 16 + g*8 + (e-8)
            const v8h* bp = (const v8h*)&bt[cur][m][kc * 32 + g * 8];
            v8h b0 = bp[0];
            v8h b1 = bp[2];
            v16h bv = __builtin_shufflevector(b0, b1, 0, 1, 2, 3, 4, 5, 6, 7,
                                              8, 9, 10, 11, 12, 13, 14, 15);
            c = __builtin_amdgcn_wmma_f32_16x16x32_f16(
                    false, a[kc], false, bv, (short)0, c, false, false);
        }

        int col = nt * 16 + m;   // C tile: lane holds column m of rows r+8g
#pragma unroll
        for (int r = 0; r < 8; ++r) {
            float v = c[r];
            if (v > maxv[r]) { maxv[r] = v; maxi[r] = col; }   // '>' keeps first max
        }
        __syncthreads();   // all waves done reading bt[cur] before it is reissued
    }

    // ---- Argmax reduction across the 16 lanes sharing each row ----
#pragma unroll
    for (int r = 0; r < 8; ++r) {
        float v = maxv[r]; int i = maxi[r];
#pragma unroll
        for (int msk = 1; msk < 16; msk <<= 1) {
            float vo = __shfl_xor(v, msk, 32);
            int   io = __shfl_xor(i, msk, 32);
            if (vo > v || (vo == v && io < i)) { v = vo; i = io; }
        }
        maxi[r] = i;
    }

    // lane needs winner of row m: reg r = m&7 from lane-half (m>>3)
    int mr  = m & 7;
    int sel = maxi[0];
    if (mr == 1) sel = maxi[1];
    if (mr == 2) sel = maxi[2];
    if (mr == 3) sel = maxi[3];
    if (mr == 4) sel = maxi[4];
    if (mr == 5) sel = maxi[5];
    if (mr == 6) sel = maxi[6];
    if (mr == 7) sel = maxi[7];
    int oth  = __shfl_xor(sel, 16, 32);
    int eidx = ((m >> 3) == g) ? sel : oth;

    if (lane < 16) outidx[n0 + lane] = (float)eidx;   // indices as float

    // ---- Phase C: gather winning embedding, transposed store, loss ----
    const float* eb = emb + (size_t)eidx * DD;
    float* ob = outq + (size_t)b * DD * TT + t0 + m;
    float lsum = 0.0f;
#pragma unroll 4
    for (int i = 0; i < 128; ++i) {
        int d = 2 * i + g;
        float q  = eb[d];
        float x  = xt[w][m][d];
        float df = q - x;
        lsum += df * df;
        ob[(size_t)d * TT] = q;   // out[b][d][t] (16 lanes -> 64B contiguous)
    }
#pragma unroll
    for (int msk = 1; msk < 32; msk <<= 1) lsum += __shfl_xor(lsum, msk, 32);
    if (lane == 0) atomicAdd(acc, lsum);
}

// ---------------------------------------------------------------------------
// Kernel 3: finalize loss = (1 + 0.25) * sum / (B*T*D)
// ---------------------------------------------------------------------------
__global__ void vq_fin(const float* __restrict__ acc, float* __restrict__ outloss) {
    *outloss = 1.25f * (*acc) * (1.0f / 16777216.0f);
}

extern "C" void kernel_launch(void* const* d_in, const int* in_sizes, int n_in,
                              void* d_out, int out_size, void* d_ws, size_t ws_size,
                              hipStream_t stream) {
    (void)in_sizes; (void)n_in; (void)out_size; (void)ws_size;
    const float* inp = (const float*)d_in[0];   // [16, 256, 4096] f32
    const float* emb = (const float*)d_in[1];   // [1024, 256] f32

    _Float16* enorm = (_Float16*)d_ws;                                   // 512 KB
    float*    acc   = (float*)((char*)d_ws + (size_t)KK * DD * sizeof(_Float16));

    float* out     = (float*)d_out;
    float* outq    = out;                 // [16, 256, 4096]
    float* outloss = out + QELEMS;        // scalar
    float* outidx  = outloss + 1;         // [65536]

    vq_prep<<<KK / 8, 256, 0, stream>>>(emb, enorm, acc);
    vq_main<<<NN / 32, 64, 0, stream>>>(inp, emb, enorm, outq, outidx, acc);
    vq_fin<<<1, 1, 0, stream>>>(acc, outloss);
}
